// PermutationEquivariantEncoder_87660282511747
// MI455X (gfx1250) — matrix operand order repaired and checked by reading, not compiled, not run on hardware
//
#include <hip/hip_runtime.h>

typedef __attribute__((ext_vector_type(16))) _Float16 v16h;
typedef __attribute__((ext_vector_type(8)))  float    v8f;

#define NNODES 100000
#define FDIM   64

// ---------------------------------------------------------------- utilities
__global__ void k_fill(float* p, float v, int n) {
    int i = blockIdx.x * blockDim.x + threadIdx.x;
    if (i < n) p[i] = v;
}

__global__ void k_edge_deg(const int* __restrict__ dst, float* __restrict__ deg, int E) {
    int e = blockIdx.x * blockDim.x + threadIdx.x;
    if (e < E) atomicAdd(&deg[dst[e]], 1.0f);
}

__global__ void k_rsqrt_inplace(float* d, int n) {
    int i = blockIdx.x * blockDim.x + threadIdx.x;
    if (i < n) d[i] = rsqrtf(d[i]);   // deg >= 1 always (self-loop)
}

// W[k][n] (row-major 64x64) -> transposed f16 split: Wt[n*64+k] = hi/lo of W[k][n]
__global__ void k_conv_w(const float* __restrict__ W,
                         _Float16* __restrict__ Whi, _Float16* __restrict__ Wlo) {
    int t = blockIdx.x * blockDim.x + threadIdx.x;   // t = n*64 + k
    if (t >= 64 * 64) return;
    int n = t >> 6, k = t & 63;
    float w = W[k * 64 + n];
    _Float16 hi = (_Float16)w;
    Whi[t] = hi;
    Wlo[t] = (_Float16)(w - (float)hi);
}

// ------------------------------------------------- WMMA GEMM: Out = X @ W (64x64)
// One wave -> 16x64 output tile. Error-compensated f16 split (3 WMMAs per B tile).
__global__ __launch_bounds__(64) void k_gemm64_wmma(
    const float* __restrict__ X, const _Float16* __restrict__ Whi,
    const _Float16* __restrict__ Wlo, float* __restrict__ Out,
    const float* __restrict__ scale,   // per-row scale (dinv) or nullptr
    const float* __restrict__ bias,    // per-col bias or nullptr
    int relu)
{
    const int lane = threadIdx.x & 31;
    const int wave = blockIdx.x * 2 + (threadIdx.x >> 5);
    const int m0   = wave * 16;
    const int mloc = lane & 15;
    const int h    = lane >> 4;
    const int mrow = m0 + mloc;

    v8f acc[4] = {};

    const float* xrow = X + (size_t)mrow * FDIM;

    #pragma unroll
    for (int k0 = 0; k0 < FDIM; k0 += 32) {
        // A fragment (contiguous k order, same as B): lane holds K in [k0+16h, +16) contiguous
        float xv[16];
        *(float4*)(&xv[0])  = *(const float4*)(xrow + k0 + 16 * h);
        *(float4*)(&xv[4])  = *(const float4*)(xrow + k0 + 16 * h + 4);
        *(float4*)(&xv[8])  = *(const float4*)(xrow + k0 + 16 * h + 8);
        *(float4*)(&xv[12]) = *(const float4*)(xrow + k0 + 16 * h + 12);
        v16h ahi, alo;
        #pragma unroll
        for (int j = 0; j < 16; ++j) {
            _Float16 hi = (_Float16)xv[j];
            ahi[j] = hi;
            alo[j] = (_Float16)(xv[j] - (float)hi);
        }
        #pragma unroll
        for (int nb = 0; nb < 4; ++nb) {
            // B fragment (32x16): lane holds column n, K = k0+16h .. +16 contiguous in Wt[n][k]
            const int noff = (mloc + 16 * nb) * 64 + k0 + 16 * h;
            v16h bhi = *(const v16h*)(Whi + noff);
            v16h blo = *(const v16h*)(Wlo + noff);
            acc[nb] = __builtin_amdgcn_wmma_f32_16x16x32_f16(
                false, ahi, false, bhi, (short)0, acc[nb], false, false);
            acc[nb] = __builtin_amdgcn_wmma_f32_16x16x32_f16(
                false, ahi, false, blo, (short)0, acc[nb], false, false);
            acc[nb] = __builtin_amdgcn_wmma_f32_16x16x32_f16(
                false, alo, false, bhi, (short)0, acc[nb], false, false);
        }
    }

    // Epilogue: C/D layout — VGPR p, lane -> row m0 + p + 8h, col (lane&15) + 16nb
    #pragma unroll
    for (int nb = 0; nb < 4; ++nb) {
        const int col = mloc + 16 * nb;
        const float bv = bias ? bias[col] : 0.0f;
        #pragma unroll
        for (int p = 0; p < 8; ++p) {
            const int row = m0 + p + 8 * h;
            float v = acc[nb][p];
            if (scale) v *= scale[row];
            v += bv;
            if (relu) v = v > 0.0f ? v : 0.0f;
            Out[(size_t)row * FDIM + col] = v;
        }
    }
}

// ---------------------------------------------- edge scatter: agg[dst] += hs[src]
// 16 lanes per edge, float4 per lane; gathers + atomics are L2-resident (25.6MB << 192MB L2)
__global__ void k_edge_agg(const int* __restrict__ src, const int* __restrict__ dst,
                           const float* __restrict__ hs, float* __restrict__ agg, int E) {
    int t = blockIdx.x * blockDim.x + threadIdx.x;
    int e = t >> 4;
    if (e >= E) return;
    int q = (t & 15) * 4;
    int s = src[e], d = dst[e];
    float4 v = *(const float4*)(hs + (size_t)s * FDIM + q);
    float* o = agg + (size_t)d * FDIM + q;
    atomicAdd(o + 0, v.x);
    atomicAdd(o + 1, v.y);
    atomicAdd(o + 2, v.z);
    atomicAdd(o + 3, v.w);
}

// y = dinv[i] * (agg + hs) + b[j]   (self-loop term folded in via hs), optional relu
__global__ void k_finalize(const float* __restrict__ agg, const float* __restrict__ hs,
                           const float* __restrict__ dinv, const float* __restrict__ bias,
                           float* __restrict__ y, int total, int relu) {
    int t = blockIdx.x * blockDim.x + threadIdx.x;
    if (t >= total) return;
    int i = t >> 6, j = t & 63;
    float v = dinv[i] * (agg[t] + hs[t]) + bias[j];
    if (relu) v = v > 0.0f ? v : 0.0f;
    y[t] = v;
}

// ------------------------------------------------------------------- driver
extern "C" void kernel_launch(void* const* d_in, const int* in_sizes, int n_in,
                              void* d_out, int out_size, void* d_ws, size_t ws_size,
                              hipStream_t stream) {
    const float* x  = (const float*)d_in[0];
    const int*   ei = (const int*)d_in[1];
    const float* W1 = (const float*)d_in[2];
    const float* b1 = (const float*)d_in[3];
    const float* W2 = (const float*)d_in[4];
    const float* b2 = (const float*)d_in[5];
    const float* Wl = (const float*)d_in[6];
    const float* bl = (const float*)d_in[7];

    const int N = in_sizes[0] / FDIM;   // 100000
    const int E = in_sizes[1] / 2;      // 1200000
    const int* src = ei;
    const int* dst = ei + E;

    float* out  = (float*)d_out;
    float* dinv = (float*)d_ws;                       // N
    float* bufA = dinv + N;                           // N*64  (hs)
    float* bufB = bufA + (size_t)N * FDIM;            // N*64  (layer outputs)
    _Float16* wt = (_Float16*)(bufB + (size_t)N * FDIM);
    _Float16* w1hi = wt + 0 * 4096, *w1lo = wt + 1 * 4096;
    _Float16* w2hi = wt + 2 * 4096, *w2lo = wt + 3 * 4096;
    _Float16* wlhi = wt + 4 * 4096, *wllo = wt + 5 * 4096;

    const int T = 256;
    const int NF = N * FDIM;
    const int gN  = (N + T - 1) / T;
    const int gE  = (E + T - 1) / T;
    const int gNF = (NF + T - 1) / T;
    const int gE16 = (E * 16 + T - 1) / T;   // 16 lanes per edge
    const int gGemm = N / 32;                // N/16 waves, 2 waves/block (exact: 3125)

    // degrees -> dinv (in place): deg = 1 (self-loop) + indegree
    k_fill<<<gN, T, 0, stream>>>(dinv, 1.0f, N);
    k_edge_deg<<<gE, T, 0, stream>>>(dst, dinv, E);
    k_rsqrt_inplace<<<gN, T, 0, stream>>>(dinv, N);

    // transpose + f16-split weights
    k_conv_w<<<16, T, 0, stream>>>(W1, w1hi, w1lo);
    k_conv_w<<<16, T, 0, stream>>>(W2, w2hi, w2lo);
    k_conv_w<<<16, T, 0, stream>>>(Wl, wlhi, wllo);

    // ---- layer 1: hs1 = (X@W1)*dinv ; agg in d_out ; y1 -> bufB (relu)
    k_fill<<<gNF, T, 0, stream>>>(out, 0.0f, NF);
    k_gemm64_wmma<<<gGemm, 64, 0, stream>>>(x, w1hi, w1lo, bufA, dinv, nullptr, 0);
    k_edge_agg<<<gE16, T, 0, stream>>>(src, dst, bufA, out, E);
    k_finalize<<<gNF, T, 0, stream>>>(out, bufA, dinv, b1, bufB, NF, 1);

    // ---- layer 2: hs2 = (y1@W2)*dinv ; agg in d_out ; y2 -> bufB
    k_fill<<<gNF, T, 0, stream>>>(out, 0.0f, NF);
    k_gemm64_wmma<<<gGemm, 64, 0, stream>>>(bufB, w2hi, w2lo, bufA, dinv, nullptr, 0);
    k_edge_agg<<<gE16, T, 0, stream>>>(src, dst, bufA, out, E);
    k_finalize<<<gNF, T, 0, stream>>>(out, bufA, dinv, b2, bufB, NF, 0);

    // ---- final linear: d_out = y2 @ Wl + bl
    k_gemm64_wmma<<<gGemm, 64, 0, stream>>>(bufB, wlhi, wllo, out, nullptr, bl, 0);
}
